// RigidAlignNet_72885595013180
// MI455X (gfx1250) — matrix operand order repaired and checked by reading, not compiled, not run on hardware
//
#include <hip/hip_runtime.h>
#include <hip/hip_bf16.h>

typedef __attribute__((ext_vector_type(16))) _Float16 v16h;
typedef __attribute__((ext_vector_type(8)))  _Float16 v8h;
typedef __attribute__((ext_vector_type(4)))  _Float16 v4h;
typedef __attribute__((ext_vector_type(8)))  float    v8f;

union AFrag { v16h v; v8h h2[2]; _Float16 e[16]; };
union CFrag { v8f  v; float e[8]; };
union H4    { v4h  v; _Float16 e[4]; };

// ---------------------------------------------------------------------------
// Gathered spherical conv as WMMA GEMM, channel-last f16 activations.
//   y[m, o] = sum_{j,c} act[(b*N + neigh[7n+j])*CIN + c] * w[o, j*CIN+c] + bias[o]
// Branchless inner loop: rows are clamped (OOB rows computed but never stored),
// weights are zero-padded to KP = ceil(K/32)*32 so no K-tail guards are needed
// (garbage A tail elements multiply zero B rows).
// ---------------------------------------------------------------------------
template<int CIN, int NCT>
__global__ __launch_bounds__(128) void conv_wmma_kernel(
    const _Float16* __restrict__ act,   // (B*N, CIN) f16
    const int* __restrict__ neigh,      // (7*N)
    const _Float16* __restrict__ wp,    // (Cout, KP) f16, zero-padded
    const float* __restrict__ bias,     // (Cout)
    float* __restrict__ y,              // (B*N, Cout) f32, pre-norm
    int Bn, int N, int Cout)
{
    constexpr int K   = 7 * CIN;
    constexpr int KCH = (K + 31) / 32;
    constexpr int KP  = KCH * 32;

    const int lane    = threadIdx.x & 31;
    const int rowTile = blockIdx.x * 4 + (threadIdx.x >> 5);
    const int colBase = blockIdx.y * (16 * NCT) + (lane & 15);
    const int hi      = lane >> 4;

    const int m  = rowTile * 16 + (lane & 15);
    const int mc = (m < Bn) ? m : (Bn - 1);        // clamp: OOB rows never stored
    const int b  = (mc >= N) ? 1 : 0;              // B == 2
    const int n  = mc - b * N;

    // Gathered row bases for the 7 ring neighbors (32-bit offsets; max ~5.2M).
    int rbase[7];
#pragma unroll
    for (int j = 0; j < 7; ++j)
        rbase[j] = (b * N + neigh[7 * n + j]) * CIN;

    CFrag acc[NCT];
#pragma unroll
    for (int t = 0; t < NCT; ++t)
#pragma unroll
        for (int r = 0; r < 8; ++r) acc[t].e[r] = 0.0f;

    for (int kc = 0; kc < KCH; ++kc) {
        const int kbase = kc * 32;
        AFrag a;
        if (CIN >= 16) {
            // Each 8-half run = 8 contiguous channels of one neighbor: one b128.
#pragma unroll
            for (int half = 0; half < 2; ++half) {
                const int k0 = kbase + half * 16 + hi * 8;
                int j = k0 / CIN;                  // compile-time shift
                if (j > 6) j = 6;                  // K tail: garbage x zero-B
                const int c = k0 % CIN;            // compile-time mask
                a.h2[half] = *(const v8h*)(act + rbase[j] + c);
            }
        } else {
            // CIN == 3 (single chunk, K = 21): scalar, still branchless.
#pragma unroll
            for (int h = 0; h < 16; ++h) {
                const int kA = kbase + ((h & 8) << 1) + hi * 8 + (h & 7);
                int j = kA / CIN, c = kA - j * CIN;
                if (j > 6) { j = 6; c = 0; }       // tail: garbage x zero-B
                a.e[h] = act[rbase[j] + c];
            }
        }

#pragma unroll
        for (int t = 0; t < NCT; ++t) {
            const int col = colBase + t * 16;
            const _Float16* wrow = wp + col * KP + kbase + hi * 8;
            AFrag bf;
            bf.h2[0] = *(const v8h*)(wrow);
            bf.h2[1] = *(const v8h*)(wrow + 16);
            acc[t].v = __builtin_amdgcn_wmma_f32_16x16x32_f16(
                false, a.v, false, bf.v, (short)0, acc[t].v, false, false);
        }
    }

    // D: reg r, lane L -> row r + 8*hi, col L&15. Channel-last store.
#pragma unroll
    for (int t = 0; t < NCT; ++t) {
        const int col = colBase + t * 16;
        const float bb = bias[col];
#pragma unroll
        for (int r = 0; r < 8; ++r) {
            const int mm = rowTile * 16 + r + hi * 8;
            if (mm < Bn) y[(long)mm * Cout + col] = acc[t].e[r] + bb;
        }
    }
}

// ---------------------------------------------------------------------------
// Pack f32 weights (Cout, K) -> f16 (Cout, KP), zero-padded K..KP-1.
__global__ void pack_w_kernel(const float* __restrict__ w, _Float16* __restrict__ wp,
                              int K, int KP)
{
    const int col = blockIdx.y;
    const int k   = blockIdx.x * 256 + threadIdx.x;
    if (k < KP) wp[col * KP + k] = (k < K) ? (_Float16)w[col * K + k] : (_Float16)0;
}

// Input (B,3,N) f32 channel-first -> (B*N, 3) f16 channel-last.
__global__ void transpose_in_kernel(const float* __restrict__ x, _Float16* __restrict__ o,
                                    int N, int total)
{
    const int i = blockIdx.x * blockDim.x + threadIdx.x;
    if (i >= total) return;
    const int row = i / 3;
    const int c   = i - row * 3;
    const int b   = (row >= N) ? 1 : 0;
    const int n   = row - b * N;
    o[i] = (_Float16)x[((long)b * 3 + c) * N + n];
}

// ---------------------------------------------------------------------------
// GroupNorm stats over channel-last f32 y: one block per (b, group).
// Deterministic fixed-order partition + LDS tree; float4 loads (cpg >= 4).
__global__ __launch_bounds__(256) void gn_reduce_kernel(
    const float* __restrict__ y, float* __restrict__ stats,
    int Cout, int sCpg /*log2(Cout/4)*/, int N)
{
    const int b   = blockIdx.x >> 2;
    const int g   = blockIdx.x & 3;
    const int cpg = Cout >> 2;
    const long cnt  = (long)cpg * N;
    const long q4   = cnt >> 2;
    const long base = (long)b * N * Cout + (long)g * cpg;

    float s = 0.0f, ss = 0.0f;
    for (long i4 = threadIdx.x; i4 < q4; i4 += 256) {
        const long e  = i4 << 2;
        const int  n  = (int)(e >> sCpg);
        const int  cc = (int)(e & (cpg - 1));
        const float4 v = *(const float4*)(y + base + (long)n * Cout + cc);
        s  += v.x + v.y + v.z + v.w;
        ss += v.x * v.x + v.y * v.y + v.z * v.z + v.w * v.w;
    }
    __shared__ float sh1[256], sh2[256];
    sh1[threadIdx.x] = s; sh2[threadIdx.x] = ss;
    __syncthreads();
    for (int o = 128; o > 0; o >>= 1) {
        if ((int)threadIdx.x < o) {
            sh1[threadIdx.x] += sh1[threadIdx.x + o];
            sh2[threadIdx.x] += sh2[threadIdx.x + o];
        }
        __syncthreads();
    }
    if (threadIdx.x == 0) {
        const float mean = sh1[0] / (float)cnt;
        float var = sh2[0] / (float)cnt - mean * mean;
        if (var < 0.0f) var = 0.0f;
        stats[blockIdx.x * 2 + 0] = mean;
        stats[blockIdx.x * 2 + 1] = rsqrtf(var + 1e-5f);
    }
}

// GN affine + LeakyReLU(0.2): 4 elements/thread (float4 in, b64 f16 out).
// A 4-aligned run shares one (b, group) since cpg >= 4.
__global__ void gn_lrelu_kernel(
    const float* __restrict__ y, const float* __restrict__ stats,
    const float* __restrict__ gamma, const float* __restrict__ beta,
    _Float16* __restrict__ out, int sCout, int sCpg, int N, int total4)
{
    const int i4 = blockIdx.x * blockDim.x + threadIdx.x;
    if (i4 >= total4) return;
    const int e = i4 << 2;
    const int o = e & ((1 << sCout) - 1);
    const int b = ((e >> sCout) >= N) ? 1 : 0;
    const int g = o >> sCpg;
    const float mean = stats[(b * 4 + g) * 2 + 0];
    const float rstd = stats[(b * 4 + g) * 2 + 1];
    const float4 yv = *(const float4*)(y + e);
    const float4 gv = *(const float4*)(gamma + o);
    const float4 bv = *(const float4*)(beta + o);
    H4 r;
    float t;
    t = (yv.x - mean) * rstd * gv.x + bv.x; r.e[0] = (_Float16)(t > 0.f ? t : 0.2f * t);
    t = (yv.y - mean) * rstd * gv.y + bv.y; r.e[1] = (_Float16)(t > 0.f ? t : 0.2f * t);
    t = (yv.z - mean) * rstd * gv.z + bv.z; r.e[2] = (_Float16)(t > 0.f ? t : 0.2f * t);
    t = (yv.w - mean) * rstd * gv.w + bv.w; r.e[3] = (_Float16)(t > 0.f ? t : 0.2f * t);
    *(v4h*)(out + e) = r.v;
}

// Mean-pool (channel-last f16): 8 channels/thread, b128 gathers.
__global__ void pool_kernel(
    const _Float16* __restrict__ xf, const int* __restrict__ neighf,
    _Float16* __restrict__ xc, int sC /*log2 C, C>=16*/, int Nf, int Nc, int total8)
{
    const int i8 = blockIdx.x * blockDim.x + threadIdx.x;
    if (i8 >= total8) return;
    const int e   = i8 << 3;
    const int c   = e & ((1 << sC) - 1);
    const int row = e >> sC;
    const int b   = (row >= Nc) ? 1 : 0;
    const int ic  = row - b * Nc;
    float s[8];
#pragma unroll
    for (int q = 0; q < 8; ++q) s[q] = 0.0f;
#pragma unroll
    for (int j = 0; j < 7; ++j) {
        const int idx = neighf[7 * ic + j];
        const v8h v = *(const v8h*)(xf + (((long)b * Nf + idx) << sC) + c);
#pragma unroll
        for (int q = 0; q < 8; ++q) s[q] += (float)v[q];
    }
    v8h o;
#pragma unroll
    for (int q = 0; q < 8; ++q) o[q] = (_Float16)(s[q] * (1.0f / 7.0f));
    *(v8h*)(xc + e) = o;
}

// Global mean over N, then 3x256 head. One block per batch element.
__global__ __launch_bounds__(256) void head_kernel(
    const _Float16* __restrict__ x, const float* __restrict__ w,
    const float* __restrict__ bias, float* __restrict__ out, int C, int N)
{
    const int b = blockIdx.x;
    const int c = threadIdx.x;
    __shared__ float mean[256];
    float s = 0.0f;
    const _Float16* p = x + (long)b * N * C + c;
    for (int nn = 0; nn < N; ++nn) s += (float)p[(long)nn * C];
    mean[c] = s / (float)N;
    __syncthreads();
    if (c < 3) {
        float acc = bias[c];
        for (int k = 0; k < C; ++k) acc += w[c * C + k] * mean[k];
        out[b * 3 + c] = acc;
    }
}

// ---------------------------------------------------------------------------
static inline int ilog2(int v) { int s = 0; while ((1 << s) < v) ++s; return s; }

extern "C" void kernel_launch(void* const* d_in, const int* in_sizes, int n_in,
                              void* d_out, int out_size, void* d_ws, size_t ws_size,
                              hipStream_t stream)
{
    static const int NSh[5] = {163842, 40962, 10242, 2562, 642};
    static const int CH[6]  = {3, 16, 32, 64, 128, 256};

    const float* x = (const float*)d_in[0];
    const int* neigh[5];
    for (int i = 0; i < 5; ++i) neigh[i] = (const int*)d_in[1 + i];
    const float* P[42];
    for (int i = 0; i < 42; ++i) P[i] = (const float*)d_in[6 + i];
    float* out = (float*)d_out;

    // Workspace layout (BUF = 2*16*163842 elements = max activation count).
    const size_t BUF = 5242944;
    float*     yf    = (float*)d_ws;                 // f32 pre-norm conv output
    _Float16*  h0    = (_Float16*)(yf + BUF);        // f16 act ping
    _Float16*  h1    = h0 + BUF;                     // f16 act pong
    _Float16*  wp    = h1 + BUF;                     // f16 packed weights (<= 458752)
    float*     stats = (float*)(wp + 458752);

    // Input -> channel-last f16.
    {
        const int total = 2 * 3 * NSh[0];
        transpose_in_kernel<<<(total + 255) / 256, 256, 0, stream>>>(x, h0, NSh[0], total);
    }

    const _Float16* cur = h0;
    int curC = 3;

    for (int i = 0; i < 5; ++i) {
        const int N = NSh[i];
        if (i > 0) {
            const int Nf = NSh[i - 1];
            _Float16* nxt = (cur == h0) ? h1 : h0;
            const int total8 = (2 * curC * N) >> 3;
            pool_kernel<<<(total8 + 255) / 256, 256, 0, stream>>>(
                cur, neigh[i - 1], nxt, ilog2(curC), Nf, N, total8);
            cur = nxt;
        }
        for (int half = 0; half < 2; ++half) {
            const int Cin  = curC;
            const int Cout = CH[i + 1];
            const int K    = 7 * Cin;
            const int KP   = ((K + 31) / 32) * 32;
            const int Bn   = 2 * N;
            const float* W  = P[i * 8 + half * 4 + 0];
            const float* Bs = P[i * 8 + half * 4 + 1];
            const float* G  = P[i * 8 + half * 4 + 2];
            const float* T  = P[i * 8 + half * 4 + 3];

            {
                dim3 pg((KP + 255) / 256, Cout);
                pack_w_kernel<<<pg, 256, 0, stream>>>(W, wp, K, KP);
            }

            const int nct = (Cout >= 64) ? 4 : (Cout >= 32 ? 2 : 1);
            const int rowTiles = (Bn + 15) / 16;
            dim3 grid((rowTiles + 3) / 4, Cout / (16 * nct));
            if      (Cin == 3)                conv_wmma_kernel<3,   1><<<grid, 128, 0, stream>>>(cur, neigh[i], wp, Bs, yf, Bn, N, Cout);
            else if (Cin == 16 && nct == 1)   conv_wmma_kernel<16,  1><<<grid, 128, 0, stream>>>(cur, neigh[i], wp, Bs, yf, Bn, N, Cout);
            else if (Cin == 16)               conv_wmma_kernel<16,  2><<<grid, 128, 0, stream>>>(cur, neigh[i], wp, Bs, yf, Bn, N, Cout);
            else if (Cin == 32 && nct == 2)   conv_wmma_kernel<32,  2><<<grid, 128, 0, stream>>>(cur, neigh[i], wp, Bs, yf, Bn, N, Cout);
            else if (Cin == 32)               conv_wmma_kernel<32,  4><<<grid, 128, 0, stream>>>(cur, neigh[i], wp, Bs, yf, Bn, N, Cout);
            else if (Cin == 64)               conv_wmma_kernel<64,  4><<<grid, 128, 0, stream>>>(cur, neigh[i], wp, Bs, yf, Bn, N, Cout);
            else if (Cin == 128)              conv_wmma_kernel<128, 4><<<grid, 128, 0, stream>>>(cur, neigh[i], wp, Bs, yf, Bn, N, Cout);
            else                              conv_wmma_kernel<256, 4><<<grid, 128, 0, stream>>>(cur, neigh[i], wp, Bs, yf, Bn, N, Cout);

            gn_reduce_kernel<<<8, 256, 0, stream>>>(yf, stats, Cout, ilog2(Cout >> 2), N);

            _Float16* nxt = (cur == h0) ? h1 : h0;
            const int total4 = (Bn * Cout) >> 2;
            gn_lrelu_kernel<<<(total4 + 255) / 256, 256, 0, stream>>>(
                yf, stats, G, T, nxt, ilog2(Cout), ilog2(Cout >> 2), N, total4);
            cur = nxt;
            curC = Cout;
        }
    }

    head_kernel<<<2, 256, 0, stream>>>(cur, P[40], P[41], out, 256, NSh[4]);
}
